// GraphSAGE_47364899340882
// MI455X (gfx1250) — compile-verified, matrix operand order checked by reading, not run on hardware
//
#include <hip/hip_runtime.h>

#define HH 128
#define CC 10
#define EPSC 1e-5f

typedef __attribute__((ext_vector_type(16))) __bf16 v16bf;
typedef __attribute__((ext_vector_type(8)))  float  v8f;

// ---------------- utility kernels ----------------

__global__ void k_zero(float* p, long n) {
  long i = (long)blockIdx.x * blockDim.x + threadIdx.x;
  long stride = (long)gridDim.x * blockDim.x;
  for (; i < n; i += stride) p[i] = 0.f;
}

__global__ void k_deg(const int* __restrict__ dst, float* __restrict__ deg, long E) {
  long e = (long)blockIdx.x * blockDim.x + threadIdx.x;
  if (e < E) atomicAdd(&deg[dst[e]], 1.0f);
}

__global__ void k_invdeg(const float* __restrict__ deg, float* __restrict__ inv, long n) {
  long i = (long)blockIdx.x * blockDim.x + threadIdx.x;
  if (i < n) {
    float d = deg[i];
    inv[i] = 1.0f / (d > 1.0f ? d : 1.0f);
  }
}

// scatter-add of neighbor features: one wave = one edge (512B coalesced gather,
// coalesced f32 atomics into a 512B span; all L2-resident on MI455X's 192MB L2)
__global__ void k_aggregate(const int* __restrict__ src, const int* __restrict__ dst,
                            const float* __restrict__ feat, float* agg, long E) {
  long t = (long)blockIdx.x * blockDim.x + threadIdx.x;
  if (t >= E * 32) return;
  long e = t >> 5;
  int fo = (int)(t & 31) << 2;
  long s = src[e];
  long d = dst[e];
  const float4 v = *reinterpret_cast<const float4*>(feat + s * HH + fo);
  float* o = agg + d * HH + fo;
  atomicAdd(o + 0, v.x);
  atomicAdd(o + 1, v.y);
  atomicAdd(o + 2, v.z);
  atomicAdd(o + 3, v.w);
}

// Swizzle a 128x128 fp32 weight matrix (K-major, W[k][n]) into bf16 WMMA
// B-fragment order: fragment (kt,ct) holds B[kt*32..+31][ct*16..+15], each lane
// owns 16 contiguous bf16 (32B) so the GEMM reads it with one b128-pair DS load.
__global__ void k_prep_w(const float* __restrict__ W, unsigned short* __restrict__ q) {
  int t = blockIdx.x * blockDim.x + threadIdx.x;
  if (t >= HH * HH) return;
  int j     = t & 15;
  int lane  = (t >> 4) & 31;
  int fc    = t >> 9;          // 0..31 fragment index
  int kt    = fc >> 3;
  int ct    = fc & 7;
  int lhalf = lane >> 4;
  int ncol  = ct * 16 + (lane & 15);
  int kk    = (j < 8) ? (lhalf * 8 + j) : (16 + lhalf * 8 + (j - 8));
  int k     = kt * 32 + kk;
  __bf16 b  = (__bf16)W[k * HH + ncol];
  q[t] = __builtin_bit_cast(unsigned short, b);
}

// ---------------- WMMA GEMM: y = (agg*inv) @ Wl + h @ Wr + bias ----------------

__device__ __forceinline__ v16bf cvt16(const float4& p0, const float4& p1,
                                       const float4& p2, const float4& p3, float s) {
  float f[16] = {p0.x, p0.y, p0.z, p0.w, p1.x, p1.y, p1.z, p1.w,
                 p2.x, p2.y, p2.z, p2.w, p3.x, p3.y, p3.z, p3.w};
  v16bf V;
#pragma unroll
  for (int j = 0; j < 16; j++) V[j] = (__bf16)(f[j] * s);
  return V;
}

// 256 threads = 8 waves; block computes 128 rows x 128 cols.
// Each wave: 16-row strip, 8 accumulator tiles (16x16 f32), 64 WMMAs.
// Weights staged to LDS via CDNA5 async global->LDS copies (ASYNCcnt path).
// yout may alias agg (all loads of a row precede its store in program order).
__global__ __launch_bounds__(256) void k_gemm(
    const float* agg, const float* __restrict__ hin,
    const float* __restrict__ inv,
    const unsigned short* __restrict__ wlq, const unsigned short* __restrict__ wrq,
    const float* __restrict__ bias, float* yout, long n) {
  __shared__ __align__(32) uint4 lds[4096];  // 64KB: [0,2048)=Wl frags, [2048,4096)=Wr frags
  const int tid = threadIdx.x;
  const uint4* wl4 = reinterpret_cast<const uint4*>(wlq);
  const uint4* wr4 = reinterpret_cast<const uint4*>(wrq);

  // async global->LDS staging: generic shared pointer's low 32 bits are the
  // LDS byte address (ISA 10.2); bypasses VGPRs.
  const unsigned lds0 = (unsigned)(size_t)&lds[0];
#pragma unroll
  for (int i = 0; i < 8; i++) {
    const int idx = tid + i * 256;
    asm volatile("global_load_async_to_lds_b128 %0, %1, off"
                 :: "v"(lds0 + idx * 16), "v"(wl4 + idx) : "memory");
    asm volatile("global_load_async_to_lds_b128 %0, %1, off"
                 :: "v"(lds0 + 32768 + idx * 16), "v"(wr4 + idx) : "memory");
  }
  asm volatile("s_wait_asynccnt 0x0" ::: "memory");
  __syncthreads();

  const int wave  = tid >> 5;
  const int lane  = tid & 31;
  const int lhalf = lane >> 4;
  const int lmod  = lane & 15;
  const long rowbase = (long)blockIdx.x * 128 + wave * 16;

  long arow_i = rowbase + lmod;           // A-matrix row owned by this lane
  if (arow_i >= n) arow_i = n - 1;        // clamp; invalid rows never stored
  const float  s    = inv[arow_i];
  const float* arow = agg + arow_i * HH;
  const float* hrow = hin + arow_i * HH;

  v8f acc[8];
#pragma unroll
  for (int i = 0; i < 8; i++)
#pragma unroll
    for (int j = 0; j < 8; j++) acc[i][j] = 0.f;

#pragma unroll
  for (int kt = 0; kt < 4; kt++) {
    const int kb = kt * 32 + lhalf * 8;
    // issue all 8 A-operand loads before any conversion (batch the waits)
    const float4 a0 = *reinterpret_cast<const float4*>(arow + kb);
    const float4 a1 = *reinterpret_cast<const float4*>(arow + kb + 4);
    const float4 a2 = *reinterpret_cast<const float4*>(arow + kb + 16);
    const float4 a3 = *reinterpret_cast<const float4*>(arow + kb + 20);
    const float4 h0 = *reinterpret_cast<const float4*>(hrow + kb);
    const float4 h1 = *reinterpret_cast<const float4*>(hrow + kb + 4);
    const float4 h2 = *reinterpret_cast<const float4*>(hrow + kb + 16);
    const float4 h3 = *reinterpret_cast<const float4*>(hrow + kb + 20);
    const v16bf Aa = cvt16(a0, a1, a2, a3, s);     // scaled neighbor-mean operand
    const v16bf Ah = cvt16(h0, h1, h2, h3, 1.0f);  // root operand
#pragma unroll
    for (int ct = 0; ct < 8; ct++) {
      const int fo = ((kt * 8 + ct) * 32 + lane) * 2;
      v16bf Bl = *reinterpret_cast<const v16bf*>(&lds[fo]);
      v16bf Br = *reinterpret_cast<const v16bf*>(&lds[2048 + fo]);
      acc[ct] = __builtin_amdgcn_wmma_f32_16x16x32_bf16(
          false, Aa, false, Bl, (short)0, acc[ct], false, false);
      acc[ct] = __builtin_amdgcn_wmma_f32_16x16x32_bf16(
          false, Ah, false, Br, (short)0, acc[ct], false, false);
    }
  }

  // C/D layout: lane l, vgpr v -> row = (l>=16)*8 + v, col = l&15
#pragma unroll
  for (int ct = 0; ct < 8; ct++) {
    const int col = ct * 16 + lmod;
    const float bv = bias[col];
#pragma unroll
    for (int v = 0; v < 8; v++) {
      const long r = rowbase + lhalf * 8 + v;
      if (r < n) yout[r * HH + col] = acc[ct][v] + bv;
    }
  }
}

// ---------------- BatchNorm stats / apply ----------------

__global__ void k_colstats(const float* __restrict__ y, float* csum, float* csq, long n) {
  int col = threadIdx.x;               // 128 threads
  long r0 = (long)blockIdx.x * 256;
  long r1 = r0 + 256;
  if (r1 > n) r1 = n;
  float s = 0.f, q = 0.f;
  for (long r = r0; r < r1; r++) {
    float v = y[r * HH + col];
    s += v;
    q += v * v;
  }
  atomicAdd(&csum[col], s);
  atomicAdd(&csq[col], q);
}

__global__ void k_bn_relu(const float* __restrict__ y,
                          const float* __restrict__ csum, const float* __restrict__ csq,
                          const float* __restrict__ g, const float* __restrict__ bt,
                          float* __restrict__ o, long n, float rcpN) {
  long t = (long)blockIdx.x * blockDim.x + threadIdx.x;
  if (t >= n) return;
  int col = (int)(t & (HH - 1));
  float m   = csum[col] * rcpN;
  float var = csq[col] * rcpN - m * m;
  float v = g[col] * (y[t] - m) * rsqrtf(var + EPSC) + bt[col];
  o[t] = v > 0.f ? v : 0.f;
}

// ---------------- output head: relu(h @ Wlin + blin) + log_softmax ----------------

__global__ void k_head(const float* __restrict__ h, const float* __restrict__ W,
                       const float* __restrict__ b, float* __restrict__ ols,
                       float* __restrict__ orelu, long n) {
  long r = (long)blockIdx.x * blockDim.x + threadIdx.x;
  if (r >= n) return;
  const float* hr = h + r * HH;
  float lg[CC];
#pragma unroll
  for (int c = 0; c < CC; c++) lg[c] = b[c];
  for (int k = 0; k < HH; k++) {
    float hv = hr[k];
#pragma unroll
    for (int c = 0; c < CC; c++) lg[c] += hv * W[k * CC + c];
  }
#pragma unroll
  for (int c = 0; c < CC; c++) lg[c] = lg[c] > 0.f ? lg[c] : 0.f;
  float mx = lg[0];
#pragma unroll
  for (int c = 1; c < CC; c++) mx = lg[c] > mx ? lg[c] : mx;
  float se = 0.f;
#pragma unroll
  for (int c = 0; c < CC; c++) se += __expf(lg[c] - mx);
  float lse = mx + __logf(se);
#pragma unroll
  for (int c = 0; c < CC; c++) {
    ols[r * CC + c]   = lg[c] - lse;
    orelu[r * CC + c] = lg[c];
  }
}

// ---------------- launch ----------------

extern "C" void kernel_launch(void* const* d_in, const int* in_sizes, int n_in,
                              void* d_out, int out_size, void* d_ws, size_t ws_size,
                              hipStream_t stream) {
  const float* x    = (const float*)d_in[0];
  const int*   eidx = (const int*)d_in[1];
  const float* Wl0  = (const float*)d_in[2];
  const float* Wr0  = (const float*)d_in[3];
  const float* b0   = (const float*)d_in[4];
  const float* g0   = (const float*)d_in[5];
  const float* bt0  = (const float*)d_in[6];
  const float* Wl   = (const float*)d_in[7];
  const float* Wr   = (const float*)d_in[8];
  const float* bb   = (const float*)d_in[9];
  const float* gg   = (const float*)d_in[10];
  const float* btv  = (const float*)d_in[11];
  const float* Wlin = (const float*)d_in[12];
  const float* blin = (const float*)d_in[13];

  const long N = in_sizes[0] / HH;
  const long E = in_sizes[1] / 2;
  const int* src = eidx;
  const int* dst = eidx + E;

  // workspace carve-up (256B aligned regions)
  char* w = (char*)d_ws;
  auto carve = [&](size_t bytes) {
    char* p = w;
    w += (bytes + 255) & ~(size_t)255;
    return p;
  };
  float* deg  = (float*)carve(N * 4);
  float* inv  = (float*)carve(N * 4);
  float* agg  = (float*)carve((size_t)N * HH * 4);   // also GEMM output (in-place)
  float* hbuf = (float*)carve((size_t)N * HH * 4);
  float* csum = (float*)carve(HH * 4);
  float* csq  = (float*)carve(HH * 4);
  unsigned short* wq[6];
  for (int i = 0; i < 6; i++) wq[i] = (unsigned short*)carve((size_t)HH * HH * 2);

  // bf16 weight fragment prep (deterministic, re-done every call)
  k_prep_w<<<64, 256, 0, stream>>>(Wl0, wq[0]);
  k_prep_w<<<64, 256, 0, stream>>>(Wr0, wq[1]);
  k_prep_w<<<64, 256, 0, stream>>>(Wl, wq[2]);
  k_prep_w<<<64, 256, 0, stream>>>(Wr, wq[3]);
  k_prep_w<<<64, 256, 0, stream>>>(Wl + HH * HH, wq[4]);
  k_prep_w<<<64, 256, 0, stream>>>(Wr + HH * HH, wq[5]);

  k_zero<<<1024, 256, 0, stream>>>(deg, N);
  k_deg<<<(int)((E + 255) / 256), 256, 0, stream>>>(dst, deg, E);
  k_invdeg<<<(int)((N + 255) / 256), 256, 0, stream>>>(deg, inv, N);

  const int gemm_blocks = (int)((N + 127) / 128);
  const float rcpN = 1.0f / (float)N;
  const float* feat = x;
  for (int L = 0; L < 3; L++) {
    const unsigned short* wlq = wq[2 * L];
    const unsigned short* wrq = wq[2 * L + 1];
    const float* bias = (L == 0) ? b0  : bb  + (L - 1) * HH;
    const float* gam  = (L == 0) ? g0  : gg  + (L - 1) * HH;
    const float* bet  = (L == 0) ? bt0 : btv + (L - 1) * HH;

    k_zero<<<4096, 256, 0, stream>>>(agg, N * HH);
    long tagg = E * 32;
    k_aggregate<<<(int)((tagg + 255) / 256), 256, 0, stream>>>(src, dst, feat, agg, E);
    k_gemm<<<gemm_blocks, 256, 0, stream>>>(agg, feat, inv, wlq, wrq, bias, agg, N);
    k_zero<<<1, 256, 0, stream>>>(csum, HH);
    k_zero<<<1, 256, 0, stream>>>(csq, HH);
    k_colstats<<<(int)((N + 255) / 256), HH, 0, stream>>>(agg, csum, csq, N);
    k_bn_relu<<<(int)((N * HH + 255) / 256), 256, 0, stream>>>(agg, csum, csq, gam, bet,
                                                               hbuf, N * HH, rcpN);
    feat = hbuf;
  }

  float* out_ls   = (float*)d_out;
  float* out_relu = out_ls + N * CC;
  k_head<<<(int)((N + 255) / 256), 256, 0, stream>>>(hbuf, Wlin, blin, out_ls, out_relu, N);
}
